// Model_79474074845624
// MI455X (gfx1250) — matrix operand order repaired and checked by reading, not compiled, not run on hardware
//
#include <hip/hip_runtime.h>
#include <hip/hip_bf16.h>
#include <cmath>

typedef _Float16 half_t;
typedef __attribute__((ext_vector_type(16))) _Float16 v16h;
typedef __attribute__((ext_vector_type(8)))  _Float16 v8h;
typedef __attribute__((ext_vector_type(8)))  float    v8f;

#define BATCH 16
#define DM    512
#define NH    8
#define DH    64
#define DFF   2048
#define COUT  7
#define PRED  512

static inline int ceil_div(int a, int b) { return (a + b - 1) / b; }

// ---------------------------------------------------------------------------
// token conv embed (circular, kernel 3) + sinusoidal pos + mark projection
// ---------------------------------------------------------------------------
__global__ void k_embed(const float* __restrict__ x, const float* __restrict__ mark,
                        const float* __restrict__ tokw, const float* __restrict__ markw,
                        float* __restrict__ out, half_t* __restrict__ outh,
                        int L, int CIN, int MIN_) {
  int idx = blockIdx.x * blockDim.x + threadIdx.x;
  int total = BATCH * L * DM;
  if (idx >= total) return;
  int d = idx % DM;
  int l = (idx / DM) % L;
  int b = idx / (DM * L);
  float acc = 0.f;
  for (int kk = 0; kk < 3; ++kk) {
    int ls = (l + kk - 1 + L) % L;
    const float* xr = x + ((size_t)b * L + ls) * CIN;
    const float* wr = tokw + (size_t)d * CIN * 3 + kk;   // tokw[d, c, kk]
    for (int c = 0; c < CIN; ++c) acc += xr[c] * wr[c * 3];
  }
  int i2 = (d >> 1) << 1;
  float freq = __expf(-logf(10000.f) * (float)i2 / (float)DM);
  float ang = (float)l * freq;
  acc += (d & 1) ? __cosf(ang) : __sinf(ang);
  const float* mr = mark + ((size_t)b * L + l) * MIN_;
  for (int m = 0; m < MIN_; ++m) acc += mr[m] * markw[m * DM + d];
  out[idx] = acc;
  outh[idx] = (half_t)acc;
}

// ---------------------------------------------------------------------------
// Weight packing: row-major W[K,N] f32 -> f16 in WMMA-fragment order:
//   Bp[(k/32)*(N/64) + n/64][nb = (n%64)/16][lane = ((k%32)/16)*16 + n%16]
//     [e = k%16]
// so a GEMM lane reads its whole B fragment as one contiguous 32B v16h load.
// ---------------------------------------------------------------------------
__device__ __forceinline__ int pack_off(int k, int n, int N) {
  return ((((k >> 5) * (N >> 6) + (n >> 6)) * 4 + ((n & 63) >> 4)) * 512)
         + ((((k & 31) >> 4) * 16 + (n & 15)) * 16) + (k & 15);
}

__global__ void k_pack_w(const float* __restrict__ W, half_t* __restrict__ out,
                         int K, int N) {
  int idx = blockIdx.x * blockDim.x + threadIdx.x;
  if (idx >= K * N) return;
  int k = idx / N, n = idx % N;
  out[pack_off(k, n, N)] = (half_t)W[idx];
}

// distil conv weight (O=512, I=512, 3) -> per-tap packed (K=I, N=O)
__global__ void k_pack_distil(const float* __restrict__ w, half_t* __restrict__ out,
                              int kk) {
  int idx = blockIdx.x * blockDim.x + threadIdx.x;
  if (idx >= DM * DM) return;
  int k = idx / DM, n = idx % DM;              // k = input ch, n = output ch
  float val = w[((size_t)n * DM + k) * 3 + kk];
  out[pack_off(k, n, DM)] = (half_t)val;
}

// ---------------------------------------------------------------------------
// WMMA GEMM: C[M,N] = A[M,K] @ Bp (+bias[n]) (+Cin) (opt GELU); A row-major
// f16, Bp fragment-packed f16 weights, C f32, optional f16 mirror.
// Requires M%128==0, N%64==0, K%32==0.
// Block = 256 threads (8 waves). Tile 128x64, BK=32. Each wave -> 16x64 strip.
// A staged via LDS in fragment order (b128 stores / b128 fragment loads);
// B fragments read directly from packed global (32B/lane, L2/WGP$ resident).
// ---------------------------------------------------------------------------
__global__ __launch_bounds__(256)
void k_gemm(const half_t* __restrict__ A, const half_t* __restrict__ Bp,
            const float* __restrict__ bias, const float* __restrict__ Cin,
            float* __restrict__ C, half_t* __restrict__ Ch,
            int M, int N, int K, int gelu) {
  __shared__ __align__(32) half_t As[8 * 32 * 16];   // 8 wave-slots x 32 lanes x 16
  const int tid  = threadIdx.x;
  const int wave = tid >> 5;
  const int lane = tid & 31;
  const int lo   = lane & 15;
  const int hi   = lane >> 4;
  const int bm   = blockIdx.x * 128;
  const int bn   = blockIdx.y * 64;
  const int nTiles = N >> 6;
  (void)M;
  v8f acc[4] = {};

  for (int k0 = 0; k0 < K; k0 += 32) {
    // stage A: 512 chunks of 8 halves, 2 per thread
    // inverse map: tile (row,k) -> lane = ((k%32)/16)*16 + row%16,
    //              e = k%16 (8 consecutive k -> 8 consecutive e)
    #pragma unroll
    for (int t = 0; t < 2; ++t) {
      int ci  = tid + t * 256;
      int row = ci >> 2;             // 0..127
      int cc  = ci & 3;              // k = cc*8 + j
      v8h a8 = *(const v8h*)(A + (size_t)(bm + row) * K + (k0 + cc * 8));
      int lds = (((row >> 4) * 32) + ((cc >> 1) * 16) + (row & 15)) * 16 + ((cc & 1) * 8);
      *(v8h*)(As + lds) = a8;
    }
    __syncthreads();
    const v16h af = *(const v16h*)(As + (wave * 32 + lane) * 16);
    const half_t* Bt = Bp + ((size_t)(k0 >> 5) * nTiles + (bn >> 6)) * 2048;
    #pragma unroll
    for (int nb = 0; nb < 4; ++nb) {
      const v16h bf = *(const v16h*)(Bt + (nb * 32 + lane) * 16);
      acc[nb] = __builtin_amdgcn_wmma_f32_16x16x32_f16(
          false, af, false, bf, (short)0, acc[nb], false, false);
    }
    __syncthreads();
  }
  // C/D layout: lanes 0-15: N=lane, VGPR r: M=r; lanes 16-31: M=8+r
  #pragma unroll
  for (int nb = 0; nb < 4; ++nb) {
    int col = bn + nb * 16 + lo;
    #pragma unroll
    for (int r = 0; r < 8; ++r) {
      int row = bm + wave * 16 + hi * 8 + r;
      float v = acc[nb][r];
      if (bias) v += bias[col];
      if (Cin)  v += Cin[(size_t)row * N + col];
      if (gelu) v = 0.5f * v * (1.0f + erff(v * 0.70710678118f));
      C[(size_t)row * N + col] = v;
      if (Ch) Ch[(size_t)row * N + col] = (half_t)v;
    }
  }
}

// ---------------------------------------------------------------------------
// ProbSparse: sampled QK measure M = max_s(qk) - mean_s over full Lk
// ---------------------------------------------------------------------------
__global__ void k_qk_sample(const float* __restrict__ Q, const float* __restrict__ Kv,
                            float* __restrict__ Mout, int Lq, int Lk, int Upart) {
  int idx = blockIdx.x * blockDim.x + threadIdx.x;
  int total = BATCH * NH * Lq;
  if (idx >= total) return;
  int l = idx % Lq;
  int h = (idx / Lq) % NH;
  int b = idx / (Lq * NH);
  const float* q = Q + ((size_t)(b * Lq + l)) * DM + h * DH;
  float mx = -1e30f, sm = 0.f;
  for (int s = 0; s < Upart; ++s) {
    unsigned rr = (unsigned)(l * 1103515245u) ^ (unsigned)(s * 2654435761u);
    rr = rr * 1664525u + 1013904223u;
    int ki = (int)(rr % (unsigned)Lk);
    const float* kp = Kv + ((size_t)(b * Lk + ki)) * DM + h * DH;
    float dot = 0.f;
    for (int d0 = 0; d0 < DH; ++d0) dot += q[d0] * kp[d0];
    mx = fmaxf(mx, dot);
    sm += dot;
  }
  Mout[idx] = mx - sm / (float)Lk;
}

// iterative top-u argmax per (b,h); destroys Mbuf (scratch, recomputed/call)
__global__ __launch_bounds__(256)
void k_topk(float* __restrict__ Mbuf, int* __restrict__ mtop, int Lq, int u) {
  __shared__ float sv[256];
  __shared__ int   si[256];
  float* Mrow = Mbuf + (size_t)blockIdx.x * Lq;
  int t = threadIdx.x;
  for (int it = 0; it < u; ++it) {
    float best = -1e30f; int bi = 0;
    for (int i = t; i < Lq; i += 256) {
      float v = Mrow[i];
      if (v > best) { best = v; bi = i; }
    }
    sv[t] = best; si[t] = bi;
    __syncthreads();
    for (int o = 128; o > 0; o >>= 1) {
      if (t < o && sv[t + o] > sv[t]) { sv[t] = sv[t + o]; si[t] = si[t + o]; }
      __syncthreads();
    }
    if (t == 0) { mtop[(size_t)blockIdx.x * u + it] = si[0]; Mrow[si[0]] = -1e30f; }
    __syncthreads();
  }
}

// ctx init: masked -> cumsum(V) along seq; else broadcast mean(V)
__global__ void k_ctx_init(const float* __restrict__ V, float* __restrict__ ctx,
                           int Lq, int Lk, int maskFlag) {
  int idx = blockIdx.x * blockDim.x + threadIdx.x;
  if (idx >= BATCH * NH * DH) return;
  int d = idx % DH;
  int h = (idx / DH) % NH;
  int b = idx / (DH * NH);
  if (maskFlag) {
    float run = 0.f;
    for (int l = 0; l < Lq; ++l) {
      run += V[((size_t)(b * Lk + l)) * DM + h * DH + d];
      ctx[(((size_t)(b * NH + h)) * Lq + l) * DH + d] = run;
    }
  } else {
    float s = 0.f;
    for (int l = 0; l < Lk; ++l) s += V[((size_t)(b * Lk + l)) * DM + h * DH + d];
    s /= (float)Lk;
    for (int l = 0; l < Lq; ++l) ctx[(((size_t)(b * NH + h)) * Lq + l) * DH + d] = s;
  }
}

// one wave per selected query: masked softmax over Lk + attn@V scatter
__global__ __launch_bounds__(32)
void k_attn_upd(const float* __restrict__ Q, const float* __restrict__ Kv,
                const float* __restrict__ V, const int* __restrict__ mtop,
                float* __restrict__ scores, float* __restrict__ ctx,
                int Lq, int Lk, int u, int maskFlag) {
  int blk = blockIdx.x;            // (b*NH+h)*u + ui
  int ui = blk % u;
  int bh = blk / u;
  int h  = bh % NH;
  int b  = bh / NH;
  int lane = threadIdx.x;
  int row  = mtop[(size_t)bh * u + ui];
  const float* q = Q + ((size_t)(b * Lq + row)) * DM + h * DH;
  float* sc = scores + (size_t)blk * Lk;
  float scale = rsqrtf((float)DH);
  float mx = -1e30f;
  for (int kI = lane; kI < Lk; kI += 32) {
    const float* kp = Kv + ((size_t)(b * Lk + kI)) * DM + h * DH;
    float dot = 0.f;
    for (int d0 = 0; d0 < DH; ++d0) dot += q[d0] * kp[d0];
    dot *= scale;
    if (maskFlag && kI > row) dot = -1e30f;
    sc[kI] = dot;
    mx = fmaxf(mx, dot);
  }
  for (int o = 16; o > 0; o >>= 1) mx = fmaxf(mx, __shfl_xor(mx, o, 32));
  float se = 0.f;
  for (int kI = lane; kI < Lk; kI += 32) {
    float e = __expf(sc[kI] - mx);
    sc[kI] = e;
    se += e;
  }
  for (int o = 16; o > 0; o >>= 1) se += __shfl_xor(se, o, 32);
  float inv = 1.f / se;
  for (int d0 = 0; d0 < DH; ++d0) {
    float a = 0.f;
    for (int kI = lane; kI < Lk; kI += 32)
      a += sc[kI] * V[((size_t)(b * Lk + kI)) * DM + h * DH + d0];
    for (int o = 16; o > 0; o >>= 1) a += __shfl_xor(a, o, 32);
    if (lane == 0) ctx[(((size_t)(b * NH + h)) * Lq + row) * DH + d0] = a * inv;
  }
}

// (B,H,L,DH) f32 -> row-major (B*L, D) f16 for the output projection GEMM
__global__ void k_ctx_pack(const float* __restrict__ ctx, half_t* __restrict__ outh, int Lq) {
  int idx = blockIdx.x * blockDim.x + threadIdx.x;
  int total = BATCH * Lq * DM;
  if (idx >= total) return;
  int d = idx % DM;
  int l = (idx / DM) % Lq;
  int b = idx / (DM * Lq);
  int h = d / DH, dd = d % DH;
  outh[idx] = (half_t)ctx[(((size_t)(b * NH + h)) * Lq + l) * DH + dd];
}

// ---------------------------------------------------------------------------
// fused (optional residual add) + LayerNorm over D=512; 256 threads/row
// ---------------------------------------------------------------------------
__global__ __launch_bounds__(256)
void k_add_ln(const float* __restrict__ X, const float* __restrict__ Y,
              const float* __restrict__ g, const float* __restrict__ bb,
              float* __restrict__ out, half_t* __restrict__ outh, int rows) {
  __shared__ float red[256];
  int row = blockIdx.x;
  int t = threadIdx.x;
  (void)rows;
  const float* xr = X + (size_t)row * DM;
  float v0 = xr[t]       + (Y ? Y[(size_t)row * DM + t]       : 0.f);
  float v1 = xr[t + 256] + (Y ? Y[(size_t)row * DM + t + 256] : 0.f);
  red[t] = v0 + v1;
  __syncthreads();
  for (int o = 128; o > 0; o >>= 1) { if (t < o) red[t] += red[t + o]; __syncthreads(); }
  float mean = red[0] / (float)DM;
  __syncthreads();
  float d0 = v0 - mean, d1 = v1 - mean;
  red[t] = d0 * d0 + d1 * d1;
  __syncthreads();
  for (int o = 128; o > 0; o >>= 1) { if (t < o) red[t] += red[t + o]; __syncthreads(); }
  float inv = rsqrtf(red[0] / (float)DM + 1e-5f);
  float o0 = d0 * inv * g[t] + bb[t];
  float o1 = d1 * inv * g[t + 256] + bb[t + 256];
  out[(size_t)row * DM + t] = o0;
  out[(size_t)row * DM + t + 256] = o1;
  if (outh) {
    outh[(size_t)row * DM + t] = (half_t)o0;
    outh[(size_t)row * DM + t + 256] = (half_t)o1;
  }
}

// ---------------------------------------------------------------------------
// distil helpers
// ---------------------------------------------------------------------------
__global__ void k_shift_f16(const float* __restrict__ x, half_t* __restrict__ out,
                            int L, int shift) {
  int idx = blockIdx.x * blockDim.x + threadIdx.x;
  int total = BATCH * L * DM;
  if (idx >= total) return;
  int d = idx % DM;
  int l = (idx / DM) % L;
  int b = idx / (DM * L);
  int ls = (l + shift + L) % L;
  out[idx] = (half_t)x[((size_t)b * L + ls) * DM + d];
}

__global__ void k_elu_bn(float* __restrict__ y, const float* __restrict__ g,
                         const float* __restrict__ bb, int n) {
  int idx = blockIdx.x * blockDim.x + threadIdx.x;
  if (idx >= n) return;
  int d = idx % DM;
  float v = y[idx] * (g[d] * 0.9999950000374997f) + bb[d];   // 1/sqrt(1+1e-5)
  y[idx] = v > 0.f ? v : (__expf(v) - 1.f);
}

__global__ void k_maxpool(const float* __restrict__ y, float* __restrict__ out,
                          half_t* __restrict__ outh, int L) {
  int Lo = L / 2;
  int idx = blockIdx.x * blockDim.x + threadIdx.x;
  int total = BATCH * Lo * DM;
  if (idx >= total) return;
  int d  = idx % DM;
  int lo = (idx / DM) % Lo;
  int b  = idx / (DM * Lo);
  float m = -1e30f;
  for (int j = -1; j <= 1; ++j) {
    int l = 2 * lo + j;
    if (l >= 0 && l < L) m = fmaxf(m, y[((size_t)b * L + l) * DM + d]);
  }
  out[((size_t)b * Lo + lo) * DM + d] = m;
  outh[((size_t)b * Lo + lo) * DM + d] = (half_t)m;
}

// final projection 512 -> 7, last PRED rows only
__global__ void k_proj(const float* __restrict__ x, const float* __restrict__ w,
                       const float* __restrict__ bb, float* __restrict__ out, int L) {
  int idx = blockIdx.x * blockDim.x + threadIdx.x;
  int total = BATCH * PRED * COUT;
  if (idx >= total) return;
  int c = idx % COUT;
  int p = (idx / COUT) % PRED;
  int b = idx / (COUT * PRED);
  int l = L - PRED + p;
  const float* xr = x + ((size_t)b * L + l) * DM;
  float a = bb[c];
  for (int d = 0; d < DM; ++d) a += xr[d] * w[d * COUT + c];
  out[idx] = a;
}

// ---------------------------------------------------------------------------
// host-side orchestration
// ---------------------------------------------------------------------------
static int clog5(int L) {
  int c = (int)std::ceil(std::log((double)L));
  int r = 5 * c;
  return r < L ? r : L;
}

struct Scratch {
  float *q, *k, *v, *ctx, *Mbuf, *scores;
  int* mtop;
  half_t* cpack;
};

static void run_attn(const half_t* xqh, int Lq, const half_t* xkvh, int Lk,
                     const half_t* w4h, const float* b4, bool mask,
                     float* attnOut, const Scratch& S, hipStream_t st) {
  int rq = BATCH * Lq, rk = BATCH * Lk;
  k_gemm<<<dim3(rq / 128, DM / 64), 256, 0, st>>>(xqh,  w4h + (size_t)0 * DM * DM, b4 + 0 * DM, nullptr, S.q, nullptr, rq, DM, DM, 0);
  k_gemm<<<dim3(rk / 128, DM / 64), 256, 0, st>>>(xkvh, w4h + (size_t)1 * DM * DM, b4 + 1 * DM, nullptr, S.k, nullptr, rk, DM, DM, 0);
  k_gemm<<<dim3(rk / 128, DM / 64), 256, 0, st>>>(xkvh, w4h + (size_t)2 * DM * DM, b4 + 2 * DM, nullptr, S.v, nullptr, rk, DM, DM, 0);
  int u = clog5(Lq), Up = clog5(Lk);
  k_qk_sample<<<ceil_div(BATCH * NH * Lq, 256), 256, 0, st>>>(S.q, S.k, S.Mbuf, Lq, Lk, Up);
  k_topk<<<BATCH * NH, 256, 0, st>>>(S.Mbuf, S.mtop, Lq, u);
  k_ctx_init<<<ceil_div(BATCH * NH * DH, 256), 256, 0, st>>>(S.v, S.ctx, Lq, Lk, mask ? 1 : 0);
  k_attn_upd<<<BATCH * NH * u, 32, 0, st>>>(S.q, S.k, S.v, S.mtop, S.scores, S.ctx, Lq, Lk, u, mask ? 1 : 0);
  k_ctx_pack<<<ceil_div(BATCH * Lq * DM, 256), 256, 0, st>>>(S.ctx, S.cpack, Lq);
  k_gemm<<<dim3(rq / 128, DM / 64), 256, 0, st>>>(S.cpack, w4h + (size_t)3 * DM * DM, b4 + 3 * DM, nullptr, attnOut, nullptr, rq, DM, DM, 0);
}

extern "C" void kernel_launch(void* const* d_in, const int* in_sizes, int n_in,
                              void* d_out, int out_size, void* d_ws, size_t ws_size,
                              hipStream_t stream) {
  (void)in_sizes; (void)n_in; (void)out_size; (void)ws_size;
  const float* x_enc      = (const float*)d_in[0];
  const float* x_mark_enc = (const float*)d_in[1];
  const float* x_dec      = (const float*)d_in[2];
  const float* x_mark_dec = (const float*)d_in[3];
  const float* enc_tok_w  = (const float*)d_in[4];
  const float* enc_mark_w = (const float*)d_in[5];
  const float* dec_tok_w  = (const float*)d_in[6];
  const float* dec_mark_w = (const float*)d_in[7];
  const float* enc_attn_w = (const float*)d_in[8];
  const float* enc_attn_b = (const float*)d_in[9];
  const float* enc_ffn1_w = (const float*)d_in[10];
  const float* enc_ffn1_b = (const float*)d_in[11];
  const float* enc_ffn2_w = (const float*)d_in[12];
  const float* enc_ffn2_b = (const float*)d_in[13];
  const float* enc_ln_g   = (const float*)d_in[14];
  const float* enc_ln_b   = (const float*)d_in[15];
  const float* distil_w   = (const float*)d_in[16];
  const float* distil_b   = (const float*)d_in[17];
  const float* distil_bn_g= (const float*)d_in[18];
  const float* distil_bn_b= (const float*)d_in[19];
  const float* enc_norm_g = (const float*)d_in[20];
  const float* enc_norm_b = (const float*)d_in[21];
  const float* dec_self_w = (const float*)d_in[22];
  const float* dec_self_b = (const float*)d_in[23];
  const float* dec_cross_w= (const float*)d_in[24];
  const float* dec_cross_b= (const float*)d_in[25];
  const float* dec_ffn1_w = (const float*)d_in[26];
  const float* dec_ffn1_b = (const float*)d_in[27];
  const float* dec_ffn2_w = (const float*)d_in[28];
  const float* dec_ffn2_b = (const float*)d_in[29];
  const float* dec_ln_g   = (const float*)d_in[30];
  const float* dec_ln_b   = (const float*)d_in[31];
  const float* dec_norm_g = (const float*)d_in[32];
  const float* dec_norm_b = (const float*)d_in[33];
  const float* proj_w     = (const float*)d_in[34];
  const float* proj_b     = (const float*)d_in[35];

  const size_t MB = (size_t)1 << 20;
  char* wp = (char*)d_ws;
  auto alloc = [&](size_t bytes) -> void* {
    void* r = (void*)wp;
    wp += (bytes + 255) & ~(size_t)255;
    return r;
  };
  // pack B-operand weights (f32 row-major K x N) into WMMA fragment order
  auto packw = [&](const float* src, int nmat, int Kd, int Nd) -> half_t* {
    half_t* dst = (half_t*)alloc((size_t)nmat * Kd * Nd * sizeof(half_t));
    for (int i = 0; i < nmat; ++i)
      k_pack_w<<<ceil_div(Kd * Nd, 256), 256, 0, stream>>>(
          src + (size_t)i * Kd * Nd, dst + (size_t)i * Kd * Nd, Kd, Nd);
    return dst;
  };

  half_t* enc_attn_wh  = packw(enc_attn_w,  12, DM, DM);
  half_t* enc_ffn1_wh  = packw(enc_ffn1_w,   3, DM, DFF);
  half_t* enc_ffn2_wh  = packw(enc_ffn2_w,   3, DFF, DM);
  half_t* dec_self_wh  = packw(dec_self_w,   8, DM, DM);
  half_t* dec_cross_wh = packw(dec_cross_w,  8, DM, DM);
  half_t* dec_ffn1_wh  = packw(dec_ffn1_w,   2, DM, DFF);
  half_t* dec_ffn2_wh  = packw(dec_ffn2_w,   2, DFF, DM);
  half_t* distil_wh    = (half_t*)alloc((size_t)2 * 3 * DM * DM * sizeof(half_t));
  for (int l = 0; l < 2; ++l)
    for (int kk = 0; kk < 3; ++kk)
      k_pack_distil<<<ceil_div(DM * DM, 256), 256, 0, stream>>>(
          distil_w + (size_t)l * DM * DM * 3,
          distil_wh + (size_t)(l * 3 + kk) * DM * DM, kk);

  // big arenas (aliased)
  char* regionA = (char*)alloc(256 * MB);   // q|k|v|ctx  ALIAS ffn-mid f32 / conv y
  char* regionB = (char*)alloc(128 * MB);   // cpack|shift|scores ALIAS ffn-mid f16
  float*  xf      = (float*)alloc(64 * MB);
  half_t* xh      = (half_t*)alloc(32 * MB);
  float*  hf      = (float*)alloc(64 * MB);
  half_t* hh      = (half_t*)alloc(32 * MB);
  float*  attnOut = (float*)alloc(64 * MB); // also FFN2 output (y2)
  float*  Mbuf    = (float*)alloc(2 * MB);
  int*    mtop    = (int*)alloc(64 * 1024);
  float*  encOut  = (float*)alloc(8 * MB);
  half_t* encOutH = (half_t*)alloc(4 * MB);

  Scratch S;
  S.q      = (float*)(regionA);
  S.k      = (float*)(regionA + 64 * MB);
  S.v      = (float*)(regionA + 128 * MB);
  S.ctx    = (float*)(regionA + 192 * MB);
  S.Mbuf   = Mbuf;
  S.mtop   = mtop;
  S.cpack  = (half_t*)(regionB);
  S.scores = (float*)(regionB + 64 * MB);
  float*  ffnMidF = (float*)regionA;        // free after attention each layer
  half_t* ffnMidH = (half_t*)regionB;
  float*  convY   = (float*)regionA;        // free during distil
  half_t* shiftH  = (half_t*)(regionB + 32 * MB);

  // ---------------- encoder ----------------
  int L = 2048;
  k_embed<<<ceil_div(BATCH * L * DM, 256), 256, 0, stream>>>(
      x_enc, x_mark_enc, enc_tok_w, enc_mark_w, xf, xh, L, 7, 4);
  for (int l = 0; l < 3; ++l) {
    int rows = BATCH * L;
    run_attn(xh, L, xh, L, enc_attn_wh + (size_t)l * 4 * DM * DM,
             enc_attn_b + (size_t)l * 4 * DM, false, attnOut, S, stream);
    k_add_ln<<<rows, 256, 0, stream>>>(xf, attnOut,
        enc_ln_g + (size_t)(l * 2 + 0) * DM, enc_ln_b + (size_t)(l * 2 + 0) * DM,
        hf, hh, rows);
    k_gemm<<<dim3(rows / 128, DFF / 64), 256, 0, stream>>>(
        hh, enc_ffn1_wh + (size_t)l * DM * DFF, enc_ffn1_b + (size_t)l * DFF,
        nullptr, ffnMidF, ffnMidH, rows, DFF, DM, 1);
    k_gemm<<<dim3(rows / 128, DM / 64), 256, 0, stream>>>(
        ffnMidH, enc_ffn2_wh + (size_t)l * DFF * DM, enc_ffn2_b + (size_t)l * DM,
        nullptr, attnOut, nullptr, rows, DM, DFF, 0);
    k_add_ln<<<rows, 256, 0, stream>>>(hf, attnOut,
        enc_ln_g + (size_t)(l * 2 + 1) * DM, enc_ln_b + (size_t)(l * 2 + 1) * DM,
        xf, xh, rows);
    if (l < 2) {
      for (int kk = 0; kk < 3; ++kk) {
        k_shift_f16<<<ceil_div(rows * DM, 256), 256, 0, stream>>>(xf, shiftH, L, kk - 1);
        k_gemm<<<dim3(rows / 128, DM / 64), 256, 0, stream>>>(
            shiftH, distil_wh + (size_t)(l * 3 + kk) * DM * DM,
            kk == 0 ? distil_b + (size_t)l * DM : nullptr,
            kk == 0 ? nullptr : convY, convY, nullptr, rows, DM, DM, 0);
      }
      k_elu_bn<<<ceil_div(rows * DM, 256), 256, 0, stream>>>(
          convY, distil_bn_g + (size_t)l * DM, distil_bn_b + (size_t)l * DM, rows * DM);
      k_maxpool<<<ceil_div(BATCH * (L / 2) * DM, 256), 256, 0, stream>>>(convY, xf, xh, L);
      L /= 2;
    }
  }
  int Lenc = L;   // 512
  k_add_ln<<<BATCH * Lenc, 256, 0, stream>>>(xf, nullptr, enc_norm_g, enc_norm_b,
                                             encOut, encOutH, BATCH * Lenc);

  // ---------------- decoder ----------------
  int Ld = 1024;
  int rows = BATCH * Ld;
  k_embed<<<ceil_div(rows * DM, 256), 256, 0, stream>>>(
      x_dec, x_mark_dec, dec_tok_w, dec_mark_w, xf, xh, Ld, 7, 4);
  for (int l = 0; l < 2; ++l) {
    run_attn(xh, Ld, xh, Ld, dec_self_wh + (size_t)l * 4 * DM * DM,
             dec_self_b + (size_t)l * 4 * DM, true, attnOut, S, stream);
    k_add_ln<<<rows, 256, 0, stream>>>(xf, attnOut,
        dec_ln_g + (size_t)(l * 3 + 0) * DM, dec_ln_b + (size_t)(l * 3 + 0) * DM,
        hf, hh, rows);
    run_attn(hh, Ld, encOutH, Lenc, dec_cross_wh + (size_t)l * 4 * DM * DM,
             dec_cross_b + (size_t)l * 4 * DM, false, attnOut, S, stream);
    k_add_ln<<<rows, 256, 0, stream>>>(hf, attnOut,
        dec_ln_g + (size_t)(l * 3 + 1) * DM, dec_ln_b + (size_t)(l * 3 + 1) * DM,
        hf, hh, rows);
    k_gemm<<<dim3(rows / 128, DFF / 64), 256, 0, stream>>>(
        hh, dec_ffn1_wh + (size_t)l * DM * DFF, dec_ffn1_b + (size_t)l * DFF,
        nullptr, ffnMidF, ffnMidH, rows, DFF, DM, 1);
    k_gemm<<<dim3(rows / 128, DM / 64), 256, 0, stream>>>(
        ffnMidH, dec_ffn2_wh + (size_t)l * DFF * DM, dec_ffn2_b + (size_t)l * DM,
        nullptr, attnOut, nullptr, rows, DM, DFF, 0);
    k_add_ln<<<rows, 256, 0, stream>>>(hf, attnOut,
        dec_ln_g + (size_t)(l * 3 + 2) * DM, dec_ln_b + (size_t)(l * 3 + 2) * DM,
        xf, xh, rows);
  }
  k_add_ln<<<rows, 256, 0, stream>>>(xf, nullptr, dec_norm_g, dec_norm_b,
                                     hf, nullptr, rows);
  k_proj<<<ceil_div(BATCH * PRED * COUT, 256), 256, 0, stream>>>(
      hf, proj_w, proj_b, (float*)d_out, Ld);
}
